// MHSA_68891275428192
// MI455X (gfx1250) — matrix operand order repaired and checked by reading, not compiled, not run on hardware
//
#include <hip/hip_runtime.h>
#include <hip/hip_bf16.h>

#define BB 4
#define SS 2048
#define DD 2048
#define HH 16
#define HD 128

typedef __attribute__((ext_vector_type(16))) __bf16 v16bf;
typedef __attribute__((ext_vector_type(8)))  __bf16 v8bf;
typedef __attribute__((ext_vector_type(8)))  float  v8f;
typedef __attribute__((ext_vector_type(4)))  int    v4i;

// ---- CDNA5 async global->LDS path (guarded: falls back if builtin absent) ----
#if defined(__has_builtin)
#if __has_builtin(__builtin_amdgcn_global_load_async_to_lds_b128)
#define HAVE_ASYNC_LDS 1
#endif
#endif

// param types per compiler diagnostic: (int4 __device__*, int4 __shared__*, int, int)
#define ASYNC_GPTR(p) ((__attribute__((address_space(1))) v4i*)(p))
#define ASYNC_LPTR(p) ((__attribute__((address_space(3))) v4i*)(p))

__device__ __forceinline__ void wait_async0() {
#if defined(HAVE_ASYNC_LDS)
#if __has_builtin(__builtin_amdgcn_s_wait_asynccnt)
    __builtin_amdgcn_s_wait_asynccnt(0);
#else
    asm volatile("s_wait_asynccnt 0x0" ::: "memory");
#endif
#endif
}

// ---------------------------------------------------------------------------
// Tiled GEMM: C[M][N] = A[M][K] * W[N][K]^T.
// Block: 256 threads = 8 waves arranged 4(M) x 2(N); block tile 256x128.
// Wave tile 64x64: 4 A-frags x 4 B-frags = 16 WMMA per 32-wide K step
// (4 ds_load per wmma).  Double-buffered LDS: next slab's global loads are
// issued before the WMMA block so HBM latency overlaps the matrix pipe.
// ---------------------------------------------------------------------------
template<bool A_BF16, bool OUT_BF16>
__global__ __launch_bounds__(256)
void gemm_bf16_wmma(const void* __restrict__ A_, const float* __restrict__ W,
                    void* __restrict__ C_, int M, int N, int K) {
    const int tid  = threadIdx.x;
    const int lane = tid & 31;
    const int wv   = tid >> 5;
    const int l    = lane & 15;
    const int hi   = lane >> 4;
    const int wm   = wv & 3;            // M quadrant (0..3) -> wm*64
    const int wn   = wv >> 2;           // N half     (0..1) -> wn*64
    const int bm   = blockIdx.y * 256;
    const int bn   = blockIdx.x * 128;

    __shared__ __bf16 lA[2][256][40];   // 40-elem rows: 80B, 16B multiple
    __shared__ __bf16 lW[2][128][40];

    v8f acc[4][4];
#pragma unroll
    for (int at = 0; at < 4; ++at)
#pragma unroll
        for (int bt = 0; bt < 4; ++bt)
            acc[at][bt] = (v8f){0.f,0.f,0.f,0.f,0.f,0.f,0.f,0.f};

    // staging map: A -> one full 32-col row per thread (row = tid);
    //              W -> half row per thread (row = tid>>1, col = (tid&1)*16)
    const int wrow = tid >> 1;
    const int wcol = (tid & 1) * 16;

    // ---- initial stage into buffer 0 ----
    {
        if (A_BF16) {
            const __bf16* src = (const __bf16*)A_ + (size_t)(bm + tid) * K;
#pragma unroll
            for (int j = 0; j < 32; ++j) lA[0][tid][j] = src[j];
        } else {
            const float* src = (const float*)A_ + (size_t)(bm + tid) * K;
#pragma unroll
            for (int j = 0; j < 32; ++j) lA[0][tid][j] = (__bf16)src[j];
        }
        const float* src = W + (size_t)(bn + wrow) * K + wcol;
#pragma unroll
        for (int j = 0; j < 16; ++j) lW[0][wrow][wcol + j] = (__bf16)src[j];
    }
    __syncthreads();

    int buf = 0;
    for (int kc = 0; kc < K; kc += 32) {
        const bool more = (kc + 32) < K;

        // ---- issue next-slab global loads into registers (overlap wmma) ----
        float  nAf[32];
        __bf16 nAb[32];
        float  nW[16];
        if (more) {
            if (A_BF16) {
                const __bf16* src = (const __bf16*)A_ + (size_t)(bm + tid) * K + kc + 32;
#pragma unroll
                for (int j = 0; j < 32; ++j) nAb[j] = src[j];
            } else {
                const float* src = (const float*)A_ + (size_t)(bm + tid) * K + kc + 32;
#pragma unroll
                for (int j = 0; j < 32; ++j) nAf[j] = src[j];
            }
            const float* src = W + (size_t)(bn + wrow) * K + kc + 32 + wcol;
#pragma unroll
            for (int j = 0; j < 16; ++j) nW[j] = src[j];
        }

        // ---- build 4 A fragments (rows wm*64 + at*16 + l) ----
        v16bf a[4];
#pragma unroll
        for (int at = 0; at < 4; ++at) {
            const int ar = wm * 64 + at * 16 + l;
#pragma unroll
            for (int i = 0; i < 8; ++i) {
                const int g  = i >> 2;
                const int kk = g * 16 + hi * 8 + (i & 3) * 2;
                a[at][2 * i]     = lA[buf][ar][kk];
                a[at][2 * i + 1] = lA[buf][ar][kk + 1];
            }
        }
        // ---- 4 B fragments x 4 A fragments = 16 WMMA ----
#pragma unroll
        for (int bt = 0; bt < 4; ++bt) {
            v16bf b;
            const int nr = wn * 64 + bt * 16 + l;
#pragma unroll
            for (int i = 0; i < 8; ++i) {
                const int kk = (i >> 2) * 16 + hi * 8 + (i & 3) * 2;
                b[2 * i]     = lW[buf][nr][kk];
                b[2 * i + 1] = lW[buf][nr][kk + 1];
            }
#pragma unroll
            for (int at = 0; at < 4; ++at)
                acc[at][bt] = __builtin_amdgcn_wmma_f32_16x16x32_bf16(
                    false, a[at], false, b, (short)0, acc[at][bt], false, false);
        }

        // ---- store next slab into the other buffer ----
        if (more) {
            if (A_BF16) {
#pragma unroll
                for (int j = 0; j < 32; ++j) lA[buf ^ 1][tid][j] = nAb[j];
            } else {
#pragma unroll
                for (int j = 0; j < 32; ++j) lA[buf ^ 1][tid][j] = (__bf16)nAf[j];
            }
#pragma unroll
            for (int j = 0; j < 16; ++j) lW[buf ^ 1][wrow][wcol + j] = (__bf16)nW[j];
        }
        __syncthreads();
        buf ^= 1;
    }

    // ---- epilogue: C layout VGPR i -> row i + 8*hi, col = lane&15 ----
#pragma unroll
    for (int at = 0; at < 4; ++at) {
#pragma unroll
        for (int bt = 0; bt < 4; ++bt) {
#pragma unroll
            for (int i = 0; i < 8; ++i) {
                const size_t r = (size_t)(bm + wm * 64 + at * 16 + hi * 8 + i);
                const size_t c = (size_t)(bn + wn * 64 + bt * 16 + l);
                if (OUT_BF16) ((__bf16*)C_)[r * N + c] = (__bf16)acc[at][bt][i];
                else          ((float*) C_)[r * N + c] = acc[at][bt][i];
            }
        }
    }
}

// ---------------------------------------------------------------------------
// RoPE (+ optional scale), f32 -> bf16.  One thread per (even,odd) pair.
// ---------------------------------------------------------------------------
__global__ __launch_bounds__(256)
void rope_to_bf16(const float* __restrict__ X, const int* __restrict__ pos,
                  __bf16* __restrict__ Y, float scale) {
    const size_t t = (size_t)blockIdx.x * blockDim.x + threadIdx.x;
    const size_t npair = (size_t)BB * SS * DD / 2;
    if (t >= npair) return;
    const int    cp  = (int)(t % (DD / 2));   // pair column within row
    const size_t row = t / (DD / 2);          // b*S + s
    const int    s   = (int)(row % SS);
    const int    j   = cp & (HD / 2 - 1);     // pair index within head
    const float  inv = __powf(10000.0f, -(float)(2 * j) / (float)HD);
    const float  ang = (float)pos[s] * inv;
    float sn, cs;
    __sincosf(ang, &sn, &cs);
    const size_t base = row * (size_t)DD + (size_t)cp * 2;
    const float x1 = X[base], x2 = X[base + 1];
    Y[base]     = (__bf16)((x1 * cs - x2 * sn) * scale);
    Y[base + 1] = (__bf16)((x1 * sn + x2 * cs) * scale);
}

// ---------------------------------------------------------------------------
// Flash attention forward (causal). 4 waves / block; wave owns a 16-row Q
// tile of one (b,h). K-chunks of 32 keys: 2 score tiles (4 wmma each) +
// 8 PV wmma per chunk.
//  - Q/K fragments are contiguous per-lane vector loads from global;
//    all 8 K fragments for a chunk are preloaded so loads clause together.
//  - V is staged to LDS with GLOBAL_LOAD_ASYNC_TO_LDS_B128 (issued before
//    the score WMMAs so the DMA overlaps QK^T), waited with s_wait_asynccnt.
//  - P goes C-layout -> A-layout through per-wave LDS.
// ---------------------------------------------------------------------------
__global__ __launch_bounds__(128)
void attn_fwd_wmma(const __bf16* __restrict__ Q, const __bf16* __restrict__ Kc,
                   const __bf16* __restrict__ V, __bf16* __restrict__ O) {
    const int tid  = threadIdx.x;
    const int lane = tid & 31;
    const int wv   = tid >> 5;
    const int l    = lane & 15;
    const int hi   = lane >> 4;
    const int bh   = blockIdx.y;
    const int b    = bh >> 4;
    const int h    = bh & 15;
    const int q0   = blockIdx.x * 64 + wv * 16;

    const size_t headoff = (size_t)b * SS * DD + (size_t)h * HD;

    __shared__ __bf16 lV[4][32][144];   // per-wave V chunk; 288B rows (16B mult)
    __shared__ __bf16 lP[4][16][32];    // per-wave P tile  (16 q x 32 keys)

    // ---- load Q fragments: 4 x (16x32 bf16) over hd ----
    v16bf qf[4];
    {
        const __bf16* qrow = Q + headoff + (size_t)(q0 + l) * DD;
#pragma unroll
        for (int c = 0; c < 4; ++c) {
            const v8bf lo = *(const v8bf*)(qrow + c * 32 + hi * 16);
            const v8bf hg = *(const v8bf*)(qrow + c * 32 + hi * 16 + 8);
#pragma unroll
            for (int j = 0; j < 8; ++j) { qf[c][j] = lo[j]; qf[c][8 + j] = hg[j]; }
        }
    }

    float mrow[8], lrow[8];
    v8f oacc[8];
#pragma unroll
    for (int i = 0; i < 8; ++i) {
        mrow[i] = -1e30f; lrow[i] = 0.f;
        oacc[i] = (v8f){0.f,0.f,0.f,0.f,0.f,0.f,0.f,0.f};
    }

    const int kcend = (q0 + 47) >> 5;   // chunks of 32 keys, causal bound
    for (int kc = 0; kc < kcend; ++kc) {
        const int kb = kc * 32;

        // ---- prefetch next chunk's K and V rows ----
        if (kc + 1 < kcend) {
            const size_t nrow = (size_t)(kb + 32 + lane) * DD;
            __builtin_prefetch(Kc + headoff + nrow, 0, 0);
            __builtin_prefetch(Kc + headoff + nrow + 64, 0, 0);
            __builtin_prefetch(V  + headoff + nrow, 0, 0);
            __builtin_prefetch(V  + headoff + nrow + 64, 0, 0);
        }

        // ---- stage V chunk (lane -> one key row), async if available ----
        {
            const __bf16* vrow = V + headoff + (size_t)(kb + lane) * DD;
            __bf16* lrowp = &lV[wv][lane][0];
#if defined(HAVE_ASYNC_LDS)
#pragma unroll
            for (int j = 0; j < 16; ++j) {
                __builtin_amdgcn_global_load_async_to_lds_b128(
                    ASYNC_GPTR(vrow + j * 8), ASYNC_LPTR(lrowp + j * 8), 0, 0);
            }
#else
#pragma unroll
            for (int j = 0; j < 8; ++j) {
                const v16bf t = *(const v16bf*)(vrow + j * 16);
#pragma unroll
                for (int e = 0; e < 16; ++e) lrowp[j * 16 + e] = t[e];
            }
#endif
        }

        // ---- preload all K fragments for this chunk (clauses the loads) ----
        v16bf kfr[2][4];
#pragma unroll
        for (int nt = 0; nt < 2; ++nt) {
            const __bf16* krow = Kc + headoff + (size_t)(kb + nt * 16 + l) * DD;
#pragma unroll
            for (int c = 0; c < 4; ++c)
                kfr[nt][c] = *(const v16bf*)(krow + c * 32 + hi * 16);
        }

        // ---- scores: 2 tiles of 16q x 16k, K over hd=128 (4 wmma each) ----
        v8f sc[2];
#pragma unroll
        for (int nt = 0; nt < 2; ++nt) {
            v8f s = (v8f){0.f,0.f,0.f,0.f,0.f,0.f,0.f,0.f};
#pragma unroll
            for (int c = 0; c < 4; ++c)
                s = __builtin_amdgcn_wmma_f32_16x16x32_bf16(
                    false, qf[c], false, kfr[nt][c], (short)0, s, false, false);
            sc[nt] = s;
        }

        // ---- causal mask + online softmax (rows i+8*hi, cols over lanes) ----
#pragma unroll
        for (int i = 0; i < 8; ++i) {
            const int r  = q0 + hi * 8 + i;
            float s0 = (kb + l      <= r) ? sc[0][i] : -1e30f;
            float s1 = (kb + 16 + l <= r) ? sc[1][i] : -1e30f;
            float mt = fmaxf(s0, s1);
            mt = fmaxf(mt, __shfl_xor(mt, 1, 32));
            mt = fmaxf(mt, __shfl_xor(mt, 2, 32));
            mt = fmaxf(mt, __shfl_xor(mt, 4, 32));
            mt = fmaxf(mt, __shfl_xor(mt, 8, 32));
            const float mnew  = fmaxf(mrow[i], mt);
            const float alpha = __expf(mrow[i] - mnew);
            const float p0 = __expf(s0 - mnew);
            const float p1 = __expf(s1 - mnew);
            float rs = p0 + p1;
            rs += __shfl_xor(rs, 1, 32);
            rs += __shfl_xor(rs, 2, 32);
            rs += __shfl_xor(rs, 4, 32);
            rs += __shfl_xor(rs, 8, 32);
            lrow[i] = lrow[i] * alpha + rs;
            mrow[i] = mnew;
#pragma unroll
            for (int nt = 0; nt < 8; ++nt) oacc[nt][i] *= alpha;
            lP[wv][hi * 8 + i][l]      = (__bf16)p0;
            lP[wv][hi * 8 + i][16 + l] = (__bf16)p1;
        }

        // ---- P fragment (A layout, 16x32) from per-wave LDS ----
        v16bf pf;
#pragma unroll
        for (int j = 0; j < 8; ++j) {
            pf[j]     = lP[wv][l][hi * 16 + j];
            pf[8 + j] = lP[wv][l][hi * 16 + 8 + j];
        }

        // ---- ensure async V copy landed, then O += P x V (8 hd tiles) ----
        wait_async0();
#pragma unroll
        for (int nt = 0; nt < 8; ++nt) {
            v16bf vf;
#pragma unroll
            for (int i = 0; i < 8; ++i) {
                vf[2 * i]     = lV[wv][hi * 16 + 2 * i][nt * 16 + l];
                vf[2 * i + 1] = lV[wv][hi * 16 + 2 * i + 1][nt * 16 + l];
            }
            oacc[nt] = __builtin_amdgcn_wmma_f32_16x16x32_bf16(
                false, pf, false, vf, (short)0, oacc[nt], false, false);
        }
    }

    // ---- normalize + write bf16 O ----
#pragma unroll
    for (int nt = 0; nt < 8; ++nt) {
#pragma unroll
        for (int i = 0; i < 8; ++i) {
            const size_t r = (size_t)(q0 + hi * 8 + i);
            O[headoff + r * DD + nt * 16 + l] = (__bf16)(oacc[nt][i] / lrow[i]);
        }
    }
}

// ---------------------------------------------------------------------------
extern "C" void kernel_launch(void* const* d_in, const int* in_sizes, int n_in,
                              void* d_out, int out_size, void* d_ws, size_t ws_size,
                              hipStream_t stream) {
    const float* x   = (const float*)d_in[0];
    const float* wq  = (const float*)d_in[1];
    const float* wk  = (const float*)d_in[2];
    const float* wv  = (const float*)d_in[3];
    const float* wo  = (const float*)d_in[4];
    const int*   pos = (const int*)d_in[5];
    float*       out = (float*)d_out;

    const int M = BB * SS;        // 8192
    const int N = DD;             // 2048
    const int K = DD;             // 2048

    char* ws = (char*)d_ws;
    const size_t f32_sz  = (size_t)M * DD * sizeof(float);    // 64 MB
    const size_t bf16_sz = (size_t)M * DD * sizeof(__bf16);   // 32 MB
    float*  Qf = (float*)(ws);
    float*  Kf = (float*)(ws + f32_sz);
    __bf16* Qb = (__bf16*)(ws + 2 * f32_sz);
    __bf16* Kb = (__bf16*)(ws + 2 * f32_sz + bf16_sz);
    __bf16* Vb = (__bf16*)(ws + 2 * f32_sz + 2 * bf16_sz);
    __bf16* Ob = (__bf16*)(ws + 2 * f32_sz + 3 * bf16_sz);

    const dim3 gGrid(N / 128, M / 256);   // (16, 32)
    const dim3 gBlk(256);

    // 1-3: QKV projections (x @ W^T)
    gemm_bf16_wmma<false, false><<<gGrid, gBlk, 0, stream>>>(x, wq, Qf, M, N, K);
    gemm_bf16_wmma<false, false><<<gGrid, gBlk, 0, stream>>>(x, wk, Kf, M, N, K);
    gemm_bf16_wmma<false, true ><<<gGrid, gBlk, 0, stream>>>(x, wv, Vb, M, N, K);

    // 4-5: RoPE + bf16 (fold 1/sqrt(HD) into Q)
    {
        const size_t npair = (size_t)BB * SS * DD / 2;
        const int nb = (int)((npair + 255) / 256);
        rope_to_bf16<<<nb, 256, 0, stream>>>(Qf, pos, Qb, 0.08838834764831845f);
        rope_to_bf16<<<nb, 256, 0, stream>>>(Kf, pos, Kb, 1.0f);
    }

    // 6: causal flash attention
    {
        const dim3 aGrid(SS / 64, BB * HH);   // (32, 64)
        attn_fwd_wmma<<<aGrid, dim3(128), 0, stream>>>(Qb, Kb, Vb, Ob);
    }

    // 7: output projection (O @ wo^T) -> f32 out
    gemm_bf16_wmma<true, false><<<gGrid, gBlk, 0, stream>>>(Ob, wo, out, M, N, K);
}
